// DeformableTransformer_56075093016862
// MI455X (gfx1250) — matrix operand order repaired and checked by reading, not compiled, not run on hardware
//
#include <hip/hip_runtime.h>
#include <hip/hip_bf16.h>

typedef __attribute__((ext_vector_type(16))) _Float16     v16h;
typedef __attribute__((ext_vector_type(8)))  float        v8f;
typedef __attribute__((ext_vector_type(4)))  unsigned int v4u;
typedef __attribute__((ext_vector_type(8)))  int          v8i_t;
typedef __attribute__((ext_vector_type(4)))  int          v4i_t;

// ---------------- problem constants ----------------
constexpr int B_   = 16;
constexpr int LQ_  = 512;
constexpr int NH_  = 8;
constexpr int DF_  = 1024;
constexpr int LV_  = 19947;          // 100*150+50*75+25*38+13*19
constexpr int MQ_  = B_ * LQ_;       // 8192 rows for query-side GEMMs
constexpr int MV_  = B_ * LV_;       // 319152 rows for value GEMM

// ---------------- TDM descriptor issue (2-D tile, f32 elements) ----------------
// D# per CDNA5 ISA 8.3/8.4. pad: after every 32 DWORDs insert 2 DWORDs
// (pad_interval code 4, pad_amount code 1) -> even f32 LDS strides.
__device__ inline void tdm_load_tile_f32(unsigned lds_addr, const float* gptr,
                                         unsigned tdim0, unsigned tdim1,
                                         unsigned tile0, unsigned tile1,
                                         unsigned stride0_elems) {
  const unsigned long long ga = (unsigned long long)(const void*)gptr;
  v4u g0;
  g0[0] = 1u;                                   // count=1, user descriptor
  g0[1] = lds_addr;                             // LDS byte address
  g0[2] = (unsigned)ga;                         // global addr [31:0]
  g0[3] = (unsigned)(ga >> 32) | (2u << 30);    // global addr [56:32] | type=2
  v8i_t g1;
  g1[0] = (int)((2u << 16) |                    // data_size = 4 bytes
                (1u << 20) |                    // pad_enable
                (4u << 22) |                    // pad_interval: 32 DWORDs
                (1u << 25));                    // pad_amount: 2 DWORDs
  g1[1] = (int)((tdim0 & 0xFFFFu) << 16);       // tensor_dim0[15:0] @ bits 63:48
  g1[2] = (int)((tdim0 >> 16) | ((tdim1 & 0xFFFFu) << 16));
  g1[3] = (int)((tdim1 >> 16) | (tile0 << 16)); // tile_dim0 @ bits 127:112
  g1[4] = (int)tile1;                           // tile_dim1; tile_dim2 = 0
  g1[5] = (int)stride0_elems;                   // tensor_dim0_stride[31:0]
  g1[6] = 0;                                    // stride0 hi / stride1 lo
  g1[7] = 0;
  const v4i_t z4 = {0, 0, 0, 0};
#if __clang_major__ >= 23
  const v8i_t z8 = {0, 0, 0, 0, 0, 0, 0, 0};
  __builtin_amdgcn_tensor_load_to_lds(g0, g1, z4, z4, z8, 0);
#else
  __builtin_amdgcn_tensor_load_to_lds(g0, g1, z4, z4, 0);
#endif
}

// ---------------- WMMA fragment builders (f32 LDS -> f16 fragments) ----------
// A-matrix 16x32 f16 layout (ISA 7.12.2): VGPR g holds K pair at
// kb = (g/4)*16 + (lane/16)*8 + (g%4)*2, row = lane%16.
__device__ inline v16h frag_a_f32(const float* __restrict__ rowp, int lane,
                                  float scale) {
  v16h f;
  const int hb = (lane >> 4) << 3;
#pragma unroll
  for (int g = 0; g < 8; ++g) {
    const int kb = ((g >> 2) << 4) + hb + ((g & 3) << 1);
    f[2 * g]     = (_Float16)(rowp[kb] * scale);
    f[2 * g + 1] = (_Float16)(rowp[kb + 1] * scale);
  }
  return f;
}

// B-matrix 32x16 f16 layout: VGPR g holds K pair at kb = (g/4)*16 + (lane/16)*8 + (g%4)*2.
// W tile in LDS is [k][n] rows of 64 f32 with +2 DWORD pad after each 32:
// element (k, c) at k*68 + c + (c>>5)*2.
__device__ inline v16h frag_b_wtile(const float* __restrict__ base, int col,
                                    int lane) {
  v16h f;
  const int hb = (lane >> 4) << 3;
  const int cc = col + ((col >> 5) << 1);
#pragma unroll
  for (int g = 0; g < 8; ++g) {
    const int kb = ((g >> 2) << 4) + hb + ((g & 3) << 1);
    f[2 * g]     = (_Float16)base[kb * 68 + cc];
    f[2 * g + 1] = (_Float16)base[(kb + 1) * 68 + cc];
  }
  return f;
}

// ---------------- TDM-fed tiled WMMA GEMM ----------------
// C[M,N] = act(A[M,K] @ W[K,N] + bias); block 256 = 8 waves, tile 64x64,
// K step 32, double-buffered tensor_load_to_lds issued by wave 0.
template <typename OutT, bool RELU>
__global__ __launch_bounds__(256) void gemm_wmma_tdm(
    const float* __restrict__ A, const float* __restrict__ W,
    const float* __restrict__ bias, OutT* __restrict__ C,
    int M, int N, int K, int ldb, int ldc) {
  __shared__ float Abuf[2][64 * 34];   // 64 rows x 32 f32, stride 34 (pad)
  __shared__ float Wbuf[2][32 * 68];   // 32 rows x 64 f32, stride 68 (pad)

  const int tid  = threadIdx.x;
  const int lane = tid & 31;
  const int wave = tid >> 5;
  const int wm   = wave & 3;
  const int wn   = wave >> 2;
  const int m0   = blockIdx.y * 64;
  const int n0   = blockIdx.x * 64;
  const int nk   = K >> 5;
  const bool issuer = (wave == 0);

  v8f acc0 = {}; v8f acc1 = {};

  if (issuer) {   // preload tile 0
    tdm_load_tile_f32((unsigned)(size_t)&Abuf[0][0], A + (size_t)m0 * K,
                      (unsigned)K, (unsigned)(M - m0), 32u, 64u, (unsigned)K);
    tdm_load_tile_f32((unsigned)(size_t)&Wbuf[0][0], W + n0,
                      (unsigned)(N - n0), (unsigned)K, 64u, 32u, (unsigned)ldb);
  }

  for (int it = 0; it < nk; ++it) {
    const int buf = it & 1;
    __syncthreads();   // everyone done reading buf^1 from previous iteration
    if (issuer) {
      if (it + 1 < nk) {
        const int k1 = (it + 1) << 5;
        tdm_load_tile_f32((unsigned)(size_t)&Abuf[buf ^ 1][0],
                          A + (size_t)m0 * K + k1,
                          (unsigned)(K - k1), (unsigned)(M - m0), 32u, 64u,
                          (unsigned)K);
        tdm_load_tile_f32((unsigned)(size_t)&Wbuf[buf ^ 1][0],
                          W + (size_t)k1 * ldb + n0,
                          (unsigned)(N - n0), (unsigned)(K - k1), 64u, 32u,
                          (unsigned)ldb);
        __builtin_amdgcn_s_wait_tensorcnt(2);   // tile `it` complete
      } else {
        __builtin_amdgcn_s_wait_tensorcnt(0);
      }
    }
    __syncthreads();   // tile `it` visible to all waves

    const v16h af  = frag_a_f32(&Abuf[buf][(wm * 16 + (lane & 15)) * 34],
                                lane, 1.0f);
    const v16h bf0 = frag_b_wtile(&Wbuf[buf][0], wn * 32 + (lane & 15), lane);
    const v16h bf1 = frag_b_wtile(&Wbuf[buf][0], wn * 32 + 16 + (lane & 15), lane);
    acc0 = __builtin_amdgcn_wmma_f32_16x16x32_f16(false, af, false, bf0,
                                                  (short)0, acc0, false, false);
    acc1 = __builtin_amdgcn_wmma_f32_16x16x32_f16(false, af, false, bf1,
                                                  (short)0, acc1, false, false);
  }

  // epilogue: C layout lane%16 = N col, VGPR g = row (lane/16)*8 + g
  const int hi    = lane >> 4;
  const int rbase = m0 + wm * 16 + hi * 8;
#pragma unroll
  for (int nsub = 0; nsub < 2; ++nsub) {
    const v8f acc = nsub ? acc1 : acc0;
    const int col = n0 + wn * 32 + nsub * 16 + (lane & 15);
    const float bv = bias[col];
#pragma unroll
    for (int g = 0; g < 8; ++g) {
      const int row = rbase + g;
      if (row < M) {
        float v = acc[g] + bv;
        if (RELU) v = fmaxf(v, 0.f);
        C[(size_t)row * ldc + col] = (OutT)v;
      }
    }
  }
}

// ---------------- flash self-attention ----------------
// one wave per (b, h, 16-query tile); qkv rows of 768: [Q | K | V] per head.
__global__ __launch_bounds__(32) void flash_attn_kernel(
    const float* __restrict__ qkv, float* __restrict__ out) {
  __shared__ _Float16 Ps[16 * 36];
  int id = blockIdx.x;
  const int qt = id & 31; id >>= 5;       // LQ/16 = 32 tiles
  const int h  = id & 7;
  const int b  = id >> 3;
  const int lane = threadIdx.x;
  const int li = lane & 15, hi = lane >> 4;
  const float scale = 0.17677669529663687f;   // 1/sqrt(32)

  v16h qf;
  {
    const int qrow = qt * 16 + li;
    const float* qp = qkv + (size_t)(b * LQ_ + qrow) * 768 + h * 32;
#pragma unroll
    for (int g = 0; g < 8; ++g) {
      const int kb = ((g >> 2) << 4) + (hi << 3) + ((g & 3) << 1);
      qf[2 * g]     = (_Float16)(qp[kb] * scale);
      qf[2 * g + 1] = (_Float16)(qp[kb + 1] * scale);
    }
  }

  float m[8], l[8];
  v8f O0 = {}; v8f O1 = {};
#pragma unroll
  for (int g = 0; g < 8; ++g) { m[g] = -1e30f; l[g] = 0.f; }

  for (int kc = 0; kc < LQ_ / 32; ++kc) {
    v8f S0 = {}, S1 = {};
#pragma unroll
    for (int sub = 0; sub < 2; ++sub) {
      const int j = kc * 32 + sub * 16 + li;
      const float* kp = qkv + (size_t)(b * LQ_ + j) * 768 + 256 + h * 32;
      v16h kf;
#pragma unroll
      for (int g = 0; g < 8; ++g) {
        const int kb = ((g >> 2) << 4) + (hi << 3) + ((g & 3) << 1);
        kf[2 * g]     = (_Float16)kp[kb];
        kf[2 * g + 1] = (_Float16)kp[kb + 1];
      }
      if (sub == 0)
        S0 = __builtin_amdgcn_wmma_f32_16x16x32_f16(false, qf, false, kf,
                                                    (short)0, S0, false, false);
      else
        S1 = __builtin_amdgcn_wmma_f32_16x16x32_f16(false, qf, false, kf,
                                                    (short)0, S1, false, false);
    }
    // online softmax; row stats replicate across each 16-lane half
#pragma unroll
    for (int g = 0; g < 8; ++g) {
      float mloc = fmaxf(S0[g], S1[g]);
#pragma unroll
      for (int o = 1; o < 16; o <<= 1) mloc = fmaxf(mloc, __shfl_xor(mloc, o, 32));
      const float mnew = fmaxf(m[g], mloc);
      const float corr = __expf(m[g] - mnew);
      const float p0 = __expf(S0[g] - mnew);
      const float p1 = __expf(S1[g] - mnew);
      float rs = p0 + p1;
#pragma unroll
      for (int o = 1; o < 16; o <<= 1) rs += __shfl_xor(rs, o, 32);
      l[g] = l[g] * corr + rs;
      m[g] = mnew;
      O0[g] *= corr; O1[g] *= corr;
      Ps[(hi * 8 + g) * 36 + li]      = (_Float16)p0;
      Ps[(hi * 8 + g) * 36 + 16 + li] = (_Float16)p1;
    }
    __syncthreads();
    v16h pf;
    {
      const _Float16* pp = &Ps[li * 36];
      const int hb = hi << 3;
#pragma unroll
      for (int g = 0; g < 8; ++g) {
        const int kb = ((g >> 2) << 4) + hb + ((g & 3) << 1);
        pf[2 * g]     = pp[kb];
        pf[2 * g + 1] = pp[kb + 1];
      }
    }
#pragma unroll
    for (int nsub = 0; nsub < 2; ++nsub) {
      const int d = nsub * 16 + li;
      v16h vf;
#pragma unroll
      for (int g = 0; g < 8; ++g) {
        const int kb = ((g >> 2) << 4) + (hi << 3) + ((g & 3) << 1);
        vf[2 * g]     = (_Float16)qkv[(size_t)(b * LQ_ + kc * 32 + kb) * 768 + 512 + h * 32 + d];
        vf[2 * g + 1] = (_Float16)qkv[(size_t)(b * LQ_ + kc * 32 + kb + 1) * 768 + 512 + h * 32 + d];
      }
      if (nsub == 0)
        O0 = __builtin_amdgcn_wmma_f32_16x16x32_f16(false, pf, false, vf,
                                                    (short)0, O0, false, false);
      else
        O1 = __builtin_amdgcn_wmma_f32_16x16x32_f16(false, pf, false, vf,
                                                    (short)0, O1, false, false);
    }
    __syncthreads();
  }
#pragma unroll
  for (int g = 0; g < 8; ++g) {
    const int qrow = qt * 16 + hi * 8 + g;
    const float inv = 1.f / l[g];
    out[(size_t)(b * LQ_ + qrow) * 256 + h * 32 + li]      = O0[g] * inv;
    out[(size_t)(b * LQ_ + qrow) * 256 + h * 32 + 16 + li] = O1[g] * inv;
  }
}

// ---------------- deformable sampling ----------------
__global__ __launch_bounds__(256) void deform_kernel(
    const float* __restrict__ offb, const float* __restrict__ awb,
    const _Float16* __restrict__ val, const float* __restrict__ refp,
    float* __restrict__ out) {
  const int Hs[4] = {100, 50, 25, 13};
  const int Ws[4] = {150, 75, 38, 19};
  const int Ss[4] = {0, 15000, 18750, 19700};

  const int gw   = blockIdx.x * 8 + (threadIdx.x >> 5);
  const int lane = threadIdx.x & 31;
  const int h = gw & 7;
  const int q = (gw >> 3) & 511;
  const int b = gw >> 12;
  const size_t rq = (size_t)(b * LQ_ + q);

  const int j = lane & 15;
  const float logit = awb[rq * 128 + h * 16 + j];
  float mx = logit;
#pragma unroll
  for (int o = 1; o < 16; o <<= 1) mx = fmaxf(mx, __shfl_xor(mx, o, 32));
  const float e = __expf(logit - mx);
  float se = e;
#pragma unroll
  for (int o = 1; o < 16; o <<= 1) se += __shfl_xor(se, o, 32);
  const float wnorm = e / se;

  const float ox = offb[rq * 256 + h * 32 + j * 2];
  const float oy = offb[rq * 256 + h * 32 + j * 2 + 1];
  const int lvl = j >> 2;
  const float rx = refp[(rq * 4 + lvl) * 2 + 0];
  const float ry = refp[(rq * 4 + lvl) * 2 + 1];

  const int d = lane;
  float acc = 0.f;
#pragma unroll
  for (int jj = 0; jj < 16; ++jj) {
    const float aj  = __shfl(wnorm, jj, 32);
    const float oxj = __shfl(ox, jj, 32);
    const float oyj = __shfl(oy, jj, 32);
    const float rxj = __shfl(rx, jj, 32);
    const float ryj = __shfl(ry, jj, 32);
    const int lidx = jj >> 2;
    const int Hl = Hs[lidx], Wl = Ws[lidx], S = Ss[lidx];
    const float xt = rxj * (float)Wl + oxj - 0.5f;
    const float yt = ryj * (float)Hl + oyj - 0.5f;
    const float x0 = floorf(xt), y0 = floorf(yt);
#pragma unroll
    for (int corner = 0; corner < 4; ++corner) {
      const int dx = corner & 1, dy = corner >> 1;
      const float xi = x0 + (float)dx, yi = y0 + (float)dy;
      const float wgt = (1.f - fabsf(xt - xi)) * (1.f - fabsf(yt - yi));
      const bool valid = (xi >= 0.f) && (xi <= (float)(Wl - 1)) &&
                         (yi >= 0.f) && (yi <= (float)(Hl - 1));
      const int xc = min(max((int)xi, 0), Wl - 1);
      const int yc = min(max((int)yi, 0), Hl - 1);
      const int idx = yc * Wl + xc;
      const float samp = (float)val[((size_t)b * LV_ + S + idx) * 256 + h * 32 + d];
      acc += valid ? (wgt * aj * samp) : 0.f;
    }
  }
  out[rq * 256 + h * 32 + d] = acc;
}

// ---------------- fused residual add + LayerNorm ----------------
__global__ __launch_bounds__(256) void add_ln_kernel(
    const float* __restrict__ x, const float* __restrict__ y,
    const float* __restrict__ s, const float* __restrict__ bb,
    float* __restrict__ out) {
  const int row = blockIdx.x;
  const int tid = threadIdx.x;
  const float v = x[(size_t)row * 256 + tid] + y[(size_t)row * 256 + tid];
  float sum = v, sq = v * v;
#pragma unroll
  for (int o = 1; o < 32; o <<= 1) {
    sum += __shfl_xor(sum, o, 32);
    sq  += __shfl_xor(sq, o, 32);
  }
  __shared__ float ssum[8], ssq[8];
  const int wave = tid >> 5, lane = tid & 31;
  if (lane == 0) { ssum[wave] = sum; ssq[wave] = sq; }
  __syncthreads();
  if (wave == 0) {
    float a = (lane < 8) ? ssum[lane] : 0.f;
    float c = (lane < 8) ? ssq[lane] : 0.f;
#pragma unroll
    for (int o = 1; o < 8; o <<= 1) { a += __shfl_xor(a, o, 32); c += __shfl_xor(c, o, 32); }
    if (lane == 0) { ssum[0] = a; ssq[0] = c; }
  }
  __syncthreads();
  const float mean = ssum[0] * (1.f / 256.f);
  const float var  = ssq[0] * (1.f / 256.f) - mean * mean;
  const float r = rsqrtf(var + 1e-5f);
  out[(size_t)row * 256 + tid] = (v - mean) * r * s[tid] + bb[tid];
}

__global__ __launch_bounds__(256) void add_kernel(
    const float* __restrict__ a, const float* __restrict__ b,
    float* __restrict__ o, int n) {
  const int i = blockIdx.x * 256 + threadIdx.x;
  if (i < n) o[i] = a[i] + b[i];
}

// ---------------- host orchestration ----------------
extern "C" void kernel_launch(void* const* d_in, const int* in_sizes, int n_in,
                              void* d_out, int out_size, void* d_ws, size_t ws_size,
                              hipStream_t stream) {
  (void)in_sizes; (void)n_in; (void)out_size; (void)ws_size;
  const float* tgt   = (const float*)d_in[0];
  const float* qpos  = (const float*)d_in[1];
  const float* refp  = (const float*)d_in[2];
  const float* src   = (const float*)d_in[3];
  const float* w_qkv = (const float*)d_in[6];
  const float* b_qkv = (const float*)d_in[7];
  const float* w_so  = (const float*)d_in[8];
  const float* b_so  = (const float*)d_in[9];
  const float* w_off = (const float*)d_in[10];
  const float* b_off = (const float*)d_in[11];
  const float* w_aw  = (const float*)d_in[12];
  const float* b_aw  = (const float*)d_in[13];
  const float* w_val = (const float*)d_in[14];
  const float* b_val = (const float*)d_in[15];
  const float* w_out = (const float*)d_in[16];
  const float* b_out = (const float*)d_in[17];
  const float* w_ff1 = (const float*)d_in[18];
  const float* b_ff1 = (const float*)d_in[19];
  const float* w_ff2 = (const float*)d_in[20];
  const float* b_ff2 = (const float*)d_in[21];
  const float* ln1_s = (const float*)d_in[22];
  const float* ln1_b = (const float*)d_in[23];
  const float* ln2_s = (const float*)d_in[24];
  const float* ln2_b = (const float*)d_in[25];
  const float* ln3_s = (const float*)d_in[26];
  const float* ln3_b = (const float*)d_in[27];

  char* wp = (char*)d_ws;
  auto alloc = [&](size_t bytes) -> char* {
    char* p = wp;
    wp += (bytes + 255) & ~(size_t)255;
    return p;
  };
  float*    qbuf   = (float*)alloc((size_t)MQ_ * 256 * 4);
  float*    qkvbuf = (float*)alloc((size_t)MQ_ * 768 * 4);
  float*    attnbf = (float*)alloc((size_t)MQ_ * 256 * 4);
  float*    sobuf  = (float*)alloc((size_t)MQ_ * 256 * 4);
  float*    t1buf  = (float*)alloc((size_t)MQ_ * 256 * 4);
  float*    q2buf  = (float*)alloc((size_t)MQ_ * 256 * 4);
  float*    offbuf = (float*)alloc((size_t)MQ_ * 256 * 4);
  float*    awbuf  = (float*)alloc((size_t)MQ_ * 128 * 4);
  float*    dsbuf  = (float*)alloc((size_t)MQ_ * 256 * 4);
  float*    dobuf  = (float*)alloc((size_t)MQ_ * 256 * 4);
  float*    t2buf  = (float*)alloc((size_t)MQ_ * 256 * 4);
  float*    f1buf  = (float*)alloc((size_t)MQ_ * 1024 * 4);
  float*    f2buf  = (float*)alloc((size_t)MQ_ * 256 * 4);
  _Float16* valbuf = (_Float16*)alloc((size_t)MV_ * 256 * 2);

  const int n_q = MQ_ * 256;
  const dim3 blk256(256);

  // q = tgt + query_pos
  add_kernel<<<n_q / 256, blk256, 0, stream>>>(tgt, qpos, qbuf, n_q);
  // Q,K projections (cols 0..511 of w_qkv), V projection from tgt (cols 512..767)
  gemm_wmma_tdm<float, false><<<dim3(512 / 64, MQ_ / 64), blk256, 0, stream>>>(
      qbuf, w_qkv, b_qkv, qkvbuf, MQ_, 512, 256, 768, 768);
  gemm_wmma_tdm<float, false><<<dim3(256 / 64, MQ_ / 64), blk256, 0, stream>>>(
      tgt, w_qkv + 512, b_qkv + 512, qkvbuf + 512, MQ_, 256, 256, 768, 768);
  // fused flash attention
  flash_attn_kernel<<<B_ * NH_ * (LQ_ / 16), dim3(32), 0, stream>>>(qkvbuf, attnbf);
  // self-attention out projection, residual + LN(ln2)
  gemm_wmma_tdm<float, false><<<dim3(256 / 64, MQ_ / 64), blk256, 0, stream>>>(
      attnbf, w_so, b_so, sobuf, MQ_, 256, 256, 256, 256);
  add_ln_kernel<<<MQ_, blk256, 0, stream>>>(tgt, sobuf, ln2_s, ln2_b, t1buf);
  // q2 = t1 + query_pos
  add_kernel<<<n_q / 256, blk256, 0, stream>>>(t1buf, qpos, q2buf, n_q);
  // value projection (dominant GEMM), f16 output, ragged M zero-filled by TDM OOB
  gemm_wmma_tdm<_Float16, false><<<dim3(256 / 64, (MV_ + 63) / 64), blk256, 0, stream>>>(
      src, w_val, b_val, valbuf, MV_, 256, 256, 256, 256);
  // offsets & attention-weight projections
  gemm_wmma_tdm<float, false><<<dim3(256 / 64, MQ_ / 64), blk256, 0, stream>>>(
      q2buf, w_off, b_off, offbuf, MQ_, 256, 256, 256, 256);
  gemm_wmma_tdm<float, false><<<dim3(128 / 64, MQ_ / 64), blk256, 0, stream>>>(
      q2buf, w_aw, b_aw, awbuf, MQ_, 128, 256, 128, 128);
  // deformable bilinear sampling
  deform_kernel<<<MQ_ * NH_ / 8, blk256, 0, stream>>>(offbuf, awbuf, valbuf, refp, dsbuf);
  // deform out projection, residual + LN(ln1)
  gemm_wmma_tdm<float, false><<<dim3(256 / 64, MQ_ / 64), blk256, 0, stream>>>(
      dsbuf, w_out, b_out, dobuf, MQ_, 256, 256, 256, 256);
  add_ln_kernel<<<MQ_, blk256, 0, stream>>>(t1buf, dobuf, ln1_s, ln1_b, t2buf);
  // FFN
  gemm_wmma_tdm<float, true><<<dim3(DF_ / 64, MQ_ / 64), blk256, 0, stream>>>(
      t2buf, w_ff1, b_ff1, f1buf, MQ_, DF_, 256, DF_, DF_);
  gemm_wmma_tdm<float, false><<<dim3(256 / 64, MQ_ / 64), blk256, 0, stream>>>(
      f1buf, w_ff2, b_ff2, f2buf, MQ_, 256, DF_, 256, 256);
  add_ln_kernel<<<MQ_, blk256, 0, stream>>>(t2buf, f2buf, ln3_s, ln3_b, (float*)d_out);
}
